// ResidualBlockDynamic_3x3_71279277244557
// MI455X (gfx1250) — compile-verified, matrix operand order checked by reading, not run on hardware
//
#include <hip/hip_runtime.h>
#include <hip/hip_bf16.h>
#include <math.h>

// ---------------------------------------------------------------------------
// ResidualBlockDynamic 3x3 for MI455X (gfx1250, wave32, WMMA + TDM)
//
// Dominant cost: dense 3x3 conv (43.5 GFLOP) -> implicit GEMM via
// v_wmma_f32_16x16x32_f16 with K = tap*64 + ci (18 k-steps of 32).
// Each wave register-blocks 2 M-tiles (32 pixels) x 4 N-tiles (64 couts):
// 144 WMMA / wave. Weights are staged to LDS once per block by the
// Tensor Data Mover (tensor_load_to_lds + s_wait_tensorcnt), so B fragments
// come from ds_load_b128 while A fragments stream from global.
// ---------------------------------------------------------------------------

#define BATCH 16
#define CH    64
#define HH    192
#define WW    192
#define MID   12
#define HP    (HH + 2)
#define WP    (WW + 2)
#define NPIX  (HH * WW)            // 36864
#define KTOT  (CH * 9)             // 576
#define BT_BYTES (CH * KTOT * 2)   // 73728
#define EPSF  1e-10f
#define STD_SPATIAL 0.47140452079103173f  // sqrt(2)/3
#define NEG_SLOPE 0.1f

typedef _Float16     v8h   __attribute__((ext_vector_type(8)));
typedef _Float16     v16h  __attribute__((ext_vector_type(16)));
typedef float        v4f   __attribute__((ext_vector_type(4)));
typedef float        v8f   __attribute__((ext_vector_type(8)));
typedef unsigned int u32x4 __attribute__((ext_vector_type(4)));
typedef int          i32x4 __attribute__((ext_vector_type(4)));
typedef int          i32x8 __attribute__((ext_vector_type(8)));

// ---------------------------------------------------------------------------
// Kernel Z: zero the padded channels-last y buffer (halo must be 0).
// ---------------------------------------------------------------------------
__global__ void k_zero(uint4* __restrict__ p, long n16) {
    long i = (long)blockIdx.x * blockDim.x + threadIdx.x;
    long stride = (long)gridDim.x * blockDim.x;
    uint4 z = {0u, 0u, 0u, 0u};
    for (; i < n16; i += stride) p[i] = z;
}

// ---------------------------------------------------------------------------
// Kernel A: pooled[b][c] = mean over H,W of x. One block per (b,c).
// ---------------------------------------------------------------------------
__global__ void k_pool(const float* __restrict__ x, float* __restrict__ pooled) {
    __shared__ float red[256];
    int bc = blockIdx.x;                       // 0..1023
    const float* p = x + (size_t)bc * NPIX;
    float s = 0.f;
    for (int i = threadIdx.x; i < NPIX; i += 256) s += p[i];
    red[threadIdx.x] = s;
    __syncthreads();
    for (int o = 128; o > 0; o >>= 1) {
        if (threadIdx.x < o) red[threadIdx.x] += red[threadIdx.x + o];
        __syncthreads();
    }
    if (threadIdx.x == 0) pooled[bc] = red[0] * (1.0f / NPIX);
}

// ---------------------------------------------------------------------------
// Kernel P: pack conv weights Wc[co][ci][u][v] -> Bt[co][t*64+ci] as f16,
// with t = u*3+v.  Bt is row-major N x K for the WMMA B fragment.
// ---------------------------------------------------------------------------
__global__ void k_pack(const float* __restrict__ Wc, _Float16* __restrict__ Bt) {
    int id = blockIdx.x * blockDim.x + threadIdx.x;   // 64*576 = 36864
    if (id >= CH * KTOT) return;
    int co = id / KTOT;
    int k  = id % KTOT;
    int t  = k / CH;
    int ci = k % CH;
    int u = t / 3, v = t % 3;
    float w = Wc[(((size_t)co * CH + ci) * 3 + u) * 3 + v];
    Bt[(size_t)co * KTOT + t * CH + ci] = (_Float16)w;
}

// ---------------------------------------------------------------------------
// Kernel B: channel-filter MLP + filter_norm_channel.
// ---------------------------------------------------------------------------
__global__ void k_cf(const float* __restrict__ pooled,
                     const float* __restrict__ W1, const float* __restrict__ b1,
                     const float* __restrict__ W2, const float* __restrict__ b2,
                     const float* __restrict__ std_ch,
                     float* __restrict__ cf) {
    __shared__ float hsh[MID];
    int b = blockIdx.x;
    int tid = threadIdx.x;
    if (tid < MID) {
        float a = b1[tid];
        for (int c = 0; c < CH; ++c) a += pooled[b * CH + c] * W1[tid * CH + c];
        hsh[tid] = a > 0.f ? a : 0.f;
    }
    __syncthreads();
    int c = tid;   // 0..63
    float raw[9];
    float m = 0.f;
    #pragma unroll
    for (int t = 0; t < 9; ++t) {
        int row = c * 9 + t;
        float a = b2[row];
        #pragma unroll
        for (int j = 0; j < MID; ++j) a += hsh[j] * W2[row * MID + j];
        raw[t] = a;
        m += a;
    }
    m *= (1.0f / 9.0f);
    float var = 0.f;
    #pragma unroll
    for (int t = 0; t < 9; ++t) { float d = raw[t] - m; var += d * d; }
    var *= (1.0f / 8.0f);                     // ddof = 1
    float sd = sqrtf(var);
    #pragma unroll
    for (int t = 0; t < 9; ++t)
        cf[((size_t)b * CH + c) * 9 + t] =
            (raw[t] - m) / (sd + EPSF) * std_ch[c * 9 + t];
}

// ---------------------------------------------------------------------------
// Kernel C: fused sf-einsum + filter_norm_spatial + DDF multiply + leaky ReLU,
// writing y as f16 channels-last into the padded buffer.
// ---------------------------------------------------------------------------
__global__ void k_ddf(const float* __restrict__ x,
                      const float* __restrict__ Ws, const float* __restrict__ bs,
                      const float* __restrict__ cf,
                      _Float16* __restrict__ ypad) {
    __shared__ float sf_s[64 * 10];
    int b = blockIdx.z;
    int h = blockIdx.y;
    int tid = threadIdx.x;

    if (tid < 64) {
        int w = blockIdx.x * 64 + tid;
        float sf[9];
        #pragma unroll
        for (int t = 0; t < 9; ++t) sf[t] = bs[t];
        for (int c = 0; c < CH; ++c) {
            float xv = x[(((size_t)b * CH + c) * HH + h) * WW + w];
            #pragma unroll
            for (int t = 0; t < 9; ++t) sf[t] += xv * Ws[t * CH + c];
        }
        float m = 0.f;
        #pragma unroll
        for (int t = 0; t < 9; ++t) m += sf[t];
        m *= (1.0f / 9.0f);
        float var = 0.f;
        #pragma unroll
        for (int t = 0; t < 9; ++t) { float d = sf[t] - m; var += d * d; }
        var *= (1.0f / 8.0f);                 // ddof = 1
        float inv = STD_SPATIAL / (sqrtf(var) + EPSF);
        #pragma unroll
        for (int t = 0; t < 9; ++t) sf_s[tid * 10 + t] = (sf[t] - m) * inv;
    }
    __syncthreads();

    int px = tid & 63;
    int cg = tid >> 6;                        // 0..3 -> 16 channels each
    int w  = blockIdx.x * 64 + px;
    float acc[16];
    #pragma unroll
    for (int j = 0; j < 16; ++j) acc[j] = 0.f;

    #pragma unroll
    for (int t = 0; t < 9; ++t) {
        int u = t / 3, v = t % 3;
        int h2 = h + u - 1;
        int w2 = w + v - 1;
        if (h2 < 0 || h2 >= HH || w2 < 0 || w2 >= WW) continue;
        float s = sf_s[px * 10 + t];
        #pragma unroll
        for (int j = 0; j < 16; ++j) {
            int c = cg * 16 + j;
            float xv = x[(((size_t)b * CH + c) * HH + h2) * WW + w2];
            float cv = cf[((size_t)b * CH + c) * 9 + t];
            acc[j] += xv * cv * s;
        }
    }

    v8h out0, out1;
    #pragma unroll
    for (int j = 0; j < 16; ++j) {
        float o = acc[j];
        o = o >= 0.f ? o : NEG_SLOPE * o;     // leaky ReLU
        if (j < 8) out0[j] = (_Float16)o; else out1[j - 8] = (_Float16)o;
    }
    _Float16* dst = ypad + (((size_t)b * HP + (h + 1)) * WP + (w + 1)) * CH + cg * 16;
    *(v8h*)dst       = out0;
    *(v8h*)(dst + 8) = out1;
}

// ---------------------------------------------------------------------------
// Kernel D: 3x3 conv as implicit GEMM via WMMA + bias + residual add.
// Grid: (W/32, H/8, B), block 256 = 8 waves, 73,728 B dynamic LDS.
// Step 1: TDM copies packed weights Bt (64x576 f16) into LDS once per block
//         (tensor_load_to_lds, 2D descriptor, data_size=8B, tile 9216x1),
//         s_wait_tensorcnt 0, barrier.
// Step 2: wave handles row hh = by*8 + wave, pixels w0..w0+31 (2 M-tiles),
//         all 64 couts (4 N-tiles).  Per k-step: 4 global b128 (A) +
//         8 ds_load_b128 (B) feed 8 WMMAs; 144 WMMA / wave total.
// Step 3: epilogue transposes the 64x32 f32 tile through the SAME LDS
//         (weights are dead by then) for coalesced v4f residual+store.
// ---------------------------------------------------------------------------
__global__ void k_conv(const _Float16* __restrict__ ypad,
                       const _Float16* __restrict__ Bt,
                       const float* __restrict__ x,
                       const float* __restrict__ bc,
                       float* __restrict__ out) {
    extern __shared__ __align__(16) unsigned char smem[];   // 73,728 B
    _Float16* Bs   = (_Float16*)smem;          // weight stage (main loop)
    float*    ldsx = (float*)smem;             // transpose stage (epilogue)

    int b    = blockIdx.z;
    int wave = threadIdx.x >> 5;
    int lane = threadIdx.x & 31;
    int lrow = lane & 15;
    int lhi  = lane >> 4;

    int hh = blockIdx.y * 8 + wave;
    int w0 = blockIdx.x * 32;

    // ---- Stage 1: TDM weight stage into LDS -------------------------------
    if (wave == 0) {
        unsigned long long ga = (unsigned long long)(uintptr_t)Bt;
        unsigned int lds_off = (unsigned int)(uintptr_t)smem;  // LDS byte offset
        const unsigned int n8 = BT_BYTES / 8;                  // 9216 x 8B elems
        // D# group 0: count=1 | lds_addr | global_addr | type=2
        u32x4 g0;
        g0[0] = 1u;
        g0[1] = lds_off;
        g0[2] = (unsigned int)(ga & 0xffffffffu);
        g0[3] = (unsigned int)((ga >> 32) & 0x01ffffffu) | 0x80000000u;
        // D# group 1: data_size=3 (8B); tensor_dim0=tile_dim0=stride0=9216;
        // tensor_dim1=tile_dim1=1; no mask/pad/iterate/barrier.
        i32x8 g1;
        g1[0] = (int)(3u << 16);
        g1[1] = (int)((n8 & 0xffffu) << 16);        // tensor_dim0[15:0]
        g1[2] = (int)((n8 >> 16) | (1u << 16));     // tensor_dim0[31:16] | tensor_dim1[15:0]
        g1[3] = (int)((n8 & 0xffffu) << 16);        // tensor_dim1[31:16]=0 | tile_dim0
        g1[4] = 1;                                  // tile_dim1=1, tile_dim2=0
        g1[5] = (int)n8;                            // tensor_dim0_stride[31:0]
        g1[6] = (int)((n8 & 0xffffu) << 16);        // stride0[47:32]=0 | stride1[15:0]
        g1[7] = 0;                                  // stride1[47:16]
        i32x4 gz = {0, 0, 0, 0};
#if defined(__clang_major__) && (__clang_major__ >= 23)
        i32x8 gz8 = {0, 0, 0, 0, 0, 0, 0, 0};
        __builtin_amdgcn_tensor_load_to_lds(g0, g1, gz, gz, gz8, 0);
#else
        __builtin_amdgcn_tensor_load_to_lds(g0, g1, gz, gz, 0);
#endif
        __builtin_amdgcn_s_wait_tensorcnt(0);
    }
    __syncthreads();

    // ---- Stage 2: WMMA main loop ------------------------------------------
    const _Float16* yb = ypad + (size_t)b * HP * WP * CH;

    v8f acc[8];                               // [mt][nt] -> acc[mt*4+nt]
    #pragma unroll
    for (int i = 0; i < 8; ++i)
        acc[i] = (v8f){0.f, 0.f, 0.f, 0.f, 0.f, 0.f, 0.f, 0.f};

    #pragma unroll
    for (int t = 0; t < 9; ++t) {
        int u = t / 3, v = t % 3;
        const _Float16* a0 =
            yb + ((size_t)(hh + u) * WP + (w0 + lrow + v)) * CH + lhi * 8;
        const _Float16* a1 = a0 + 16 * CH;    // second M-tile: +16 pixels
        const _Float16* bp = Bs + (size_t)lrow * KTOT + t * CH + lhi * 8;
        #pragma unroll
        for (int kc = 0; kc < 2; ++kc) {      // ci chunks of 32
            v8h a0lo = *(const v8h*)(a0 + kc * 32);
            v8h a0hi = *(const v8h*)(a0 + kc * 32 + 16);
            v8h a1lo = *(const v8h*)(a1 + kc * 32);
            v8h a1hi = *(const v8h*)(a1 + kc * 32 + 16);
            v16h A0 = __builtin_shufflevector(a0lo, a0hi, 0,1,2,3,4,5,6,7,
                                              8,9,10,11,12,13,14,15);
            v16h A1 = __builtin_shufflevector(a1lo, a1hi, 0,1,2,3,4,5,6,7,
                                              8,9,10,11,12,13,14,15);
            #pragma unroll
            for (int nt = 0; nt < 4; ++nt) {
                const _Float16* bq = bp + nt * 16 * KTOT + kc * 32;
                v8h blo = *(const v8h*)bq;               // ds_load_b128
                v8h bhi = *(const v8h*)(bq + 16);        // ds_load_b128
                v16h Bf = __builtin_shufflevector(blo, bhi, 0,1,2,3,4,5,6,7,
                                                  8,9,10,11,12,13,14,15);
                acc[0 * 4 + nt] = __builtin_amdgcn_wmma_f32_16x16x32_f16(
                    false, A0, false, Bf, (short)0, acc[0 * 4 + nt], false, false);
                acc[1 * 4 + nt] = __builtin_amdgcn_wmma_f32_16x16x32_f16(
                    false, A1, false, Bf, (short)0, acc[1 * 4 + nt], false, false);
            }
        }
    }

    // ---- Stage 3: epilogue, LDS transpose (weights are dead now) ----------
    // D layout: VGPR r, lanes 0-15 -> M=r, lanes 16-31 -> M=r+8; N = lane&15.
    for (int hsel = 0; hsel < 2; ++hsel) {
        __syncthreads();
        #pragma unroll
        for (int mt = 0; mt < 2; ++mt) {
            #pragma unroll
            for (int ntl = 0; ntl < 2; ++ntl) {
                int nt = hsel * 2 + ntl;
                v8f a = acc[mt * 4 + nt];
                int col = ntl * 16 + lrow;            // co within half
                int wl  = mt * 16 + lhi * 8;          // pixel within 32
                float* p = &ldsx[(wave * 32 + col) * 36 + wl];
                *(v4f*)p       = (v4f){a[0], a[1], a[2], a[3]};
                *(v4f*)(p + 4) = (v4f){a[4], a[5], a[6], a[7]};
            }
        }
        __syncthreads();
        #pragma unroll
        for (int g = 0; g < 8; ++g) {
            int flat = g * 32 + lane;                 // 0..255
            int col  = flat >> 3;                     // 0..31
            int wl   = (flat & 7) * 4;                // 0..28
            v4f vv = *(v4f*)&ldsx[(wave * 32 + col) * 36 + wl];
            int co = hsel * 32 + col;
            float bias = bc[co];
            size_t idx = (((size_t)b * CH + co) * HH + hh) * WW + (w0 + wl);
            v4f xv = *(const v4f*)(x + idx);
            v4f o  = {xv[0] + vv[0] + bias, xv[1] + vv[1] + bias,
                      xv[2] + vv[2] + bias, xv[3] + vv[3] + bias};
            *(v4f*)(out + idx) = o;
        }
    }
}

// ---------------------------------------------------------------------------
// Host-side launcher
// ---------------------------------------------------------------------------
extern "C" void kernel_launch(void* const* d_in, const int* in_sizes, int n_in,
                              void* d_out, int out_size, void* d_ws, size_t ws_size,
                              hipStream_t stream) {
    const float* x      = (const float*)d_in[0];
    const float* Ws     = (const float*)d_in[1];
    const float* bs     = (const float*)d_in[2];
    const float* W1     = (const float*)d_in[3];
    const float* b1     = (const float*)d_in[4];
    const float* W2     = (const float*)d_in[5];
    const float* b2     = (const float*)d_in[6];
    const float* std_ch = (const float*)d_in[7];
    const float* Wc     = (const float*)d_in[8];
    const float* bc     = (const float*)d_in[9];
    float* out = (float*)d_out;

    // Workspace layout (bytes, 256-aligned):
    //   pooled : 16*64*4            = 4096      @ 0
    //   cf     : 16*64*9*4          = 36864     @ 4096
    //   Bt     : 64*576*2           = 73728     @ 40960
    //   ypad   : 16*194*194*64*2    = 77078528  @ 114688
    char* ws = (char*)d_ws;
    float*    pooled = (float*)(ws);
    float*    cf     = (float*)(ws + 4096);
    _Float16* Bt     = (_Float16*)(ws + 40960);
    _Float16* ypad   = (_Float16*)(ws + 114688);

    const long ypad_halves = (long)BATCH * HP * WP * CH;      // 38,539,264
    const long n16 = ypad_halves * 2 / 16;                    // 16B chunks

    k_zero<<<4096, 256, 0, stream>>>((uint4*)ypad, n16);
    k_pool<<<BATCH * CH, 256, 0, stream>>>(x, pooled);
    k_pack<<<(CH * KTOT + 255) / 256, 256, 0, stream>>>(Wc, Bt);
    k_cf<<<BATCH, 64, 0, stream>>>(pooled, W1, b1, W2, b2, std_ch, cf);
    k_ddf<<<dim3(WW / 64, HH, BATCH), 256, 0, stream>>>(x, Ws, bs, cf, ypad);
    k_conv<<<dim3(WW / 32, HH / 8, BATCH), 256, BT_BYTES, stream>>>(
        ypad, Bt, x, bc, out);
}
